// FactoredCausalSelfAttention_44135083934208
// MI455X (gfx1250) — matrix operand order repaired and checked by reading, not compiled, not run on hardware
//
#include <hip/hip_runtime.h>
#include <hip/hip_bf16.h>
#include <stdint.h>

// ---------------------------------------------------------------------------
// FactoredCausalSelfAttention for MI455X (gfx1250), wave32 + WMMA bf16.
//   B=2, T=2048, C=1024, H=16, hd=64.  All matmuls via v_wmma_f32_16x16x32_bf16.
// Workspace layout (bytes), total 50,331,648 (~48 MB):
//   [ 0        ..  8388608) x_bf16      [4096,1024]
//   [ 8388608  .. 14680064) w_attn_bf16 [3072,1024]
//   [14680064  .. 16777216) w_proj_bf16 [1024,1024]
//   [16777216  .. 25165824) Q bf16      [32,2048,64]   (bh-major)
//   [25165824  .. 33554432) K bf16      [32,2048,64]
//   [33554432  .. 41943040) Veff^T bf16 [32,64,2048]   (transposed for PV WMMA)
//   [41943040  .. 50331648) Y bf16      [4096,1024]
// ---------------------------------------------------------------------------

typedef unsigned short u16;
typedef __bf16 bf16_t;
typedef bf16_t v16bf __attribute__((ext_vector_type(16)));
typedef float v8f __attribute__((ext_vector_type(8)));
typedef unsigned int v8u __attribute__((ext_vector_type(8)));

#define BSZ 2
#define TSEQ 2048
#define CEMB 1024
#define HHEADS 16
#define MROWS (BSZ * TSEQ) /* 4096 */

__device__ __forceinline__ u16 f2bf(float f) {
  // round-to-nearest-even f32 -> bf16
  unsigned int u = __float_as_uint(f);
  u += 0x7FFFu + ((u >> 16) & 1u);
  return (u16)(u >> 16);
}

__device__ __forceinline__ v8f zero_v8f() {
  v8f z = {0.f, 0.f, 0.f, 0.f, 0.f, 0.f, 0.f, 0.f};
  return z;
}

// --------------------------- f32 -> bf16 convert ---------------------------
// 4 elements/thread: float4 in, packed uint2 (4 x bf16) out.
__global__ void cvt_bf16_kernel(const float4* __restrict__ src,
                                uint2* __restrict__ dst, int n4) {
  int i = blockIdx.x * 256 + threadIdx.x;
  if (i < n4) {
    float4 f = src[i];
    uint2 o;
    o.x = (unsigned)f2bf(f.x) | ((unsigned)f2bf(f.y) << 16);
    o.y = (unsigned)f2bf(f.z) | ((unsigned)f2bf(f.w) << 16);
    dst[i] = o;
  }
}

// ------------------------- shared WMMA GEMM tile ---------------------------
// One wave computes a 32(M) x 64(N) f32 tile of  A[M,K] * W[N,K]^T.
// Two 16-row A-fragments share each B-fragment (doubles B reuse vs 16x64).
// A-fragment (16x32 bf16): lane m=l&15 holds row m; lanes<16: K=0..15,
//   lanes>=16: K=16..31 (2 per vgpr), matching the B-fragment K order.
// B-fragment (32x16 bf16): lane holds col n=l&15; lanes<16: K=0..15,
//   lanes>=16: K=16..31 (2 per vgpr).  C/D: row = r + ((l&16)?8:0), col = l&15.
// acc[j] = rows row0..row0+15, acc[4+j] = rows row0+16..row0+31; cols n0+16j.
__device__ __forceinline__ void wmma_tile_32x64(const u16* __restrict__ A,
                                                const u16* __restrict__ W,
                                                int K, int row0, int n0,
                                                int lane, v8f acc[8]) {
  const int m   = lane & 15;
  const int akb = (lane & 16) ? 8 : 0;   // A fragment K base
  const int bkh = (lane & 16) ? 16 : 0;  // B fragment K half
  const u16* arow0 = A + (size_t)(row0 + m) * K;
  const u16* arow1 = A + (size_t)(row0 + 16 + m) * K;
  for (int k0 = 0; k0 < K; k0 += 32) {
    v8u au0, au1;
#pragma unroll
    for (int v = 0; v < 4; ++v) {
      au0[v]     = *(const unsigned int*)(arow0 + k0 + bkh + 2 * v);
      au0[v + 4] = *(const unsigned int*)(arow0 + k0 + bkh + 8 + 2 * v);
      au1[v]     = *(const unsigned int*)(arow1 + k0 + bkh + 2 * v);
      au1[v + 4] = *(const unsigned int*)(arow1 + k0 + bkh + 8 + 2 * v);
    }
    v16bf af0 = __builtin_bit_cast(v16bf, au0);
    v16bf af1 = __builtin_bit_cast(v16bf, au1);
#pragma unroll
    for (int j = 0; j < 4; ++j) {
      const u16* wrow = W + (size_t)(n0 + j * 16 + m) * K + k0 + bkh;
      v8u bu;
#pragma unroll
      for (int v = 0; v < 8; ++v)
        bu[v] = *(const unsigned int*)(wrow + 2 * v);
      v16bf bf_ = __builtin_bit_cast(v16bf, bu);
      acc[j] = __builtin_amdgcn_wmma_f32_16x16x32_bf16(
          false, af0, false, bf_, (short)0, acc[j], false, false);
      acc[4 + j] = __builtin_amdgcn_wmma_f32_16x16x32_bf16(
          false, af1, false, bf_, (short)0, acc[4 + j], false, false);
    }
  }
}

// ----------------------------- QKV projection -----------------------------
// out n<C -> Q[bh,t,d]; n<2C -> K[bh,t,d]; else Veff^T[bh,d,t] = (v+b)*xt.
__global__ __launch_bounds__(128) void qkv_gemm_kernel(
    const u16* __restrict__ xb, const u16* __restrict__ wab,
    const float* __restrict__ bias, const float* __restrict__ xt,
    u16* __restrict__ qb, u16* __restrict__ kb, u16* __restrict__ vtb) {
  const int lane = threadIdx.x & 31;
  const int wave = threadIdx.x >> 5;
  const int row0 = blockIdx.x * 32;
  const int n0 = blockIdx.y * 256 + wave * 64;
  v8f acc[8];
#pragma unroll
  for (int i = 0; i < 8; ++i) acc[i] = zero_v8f();
  wmma_tile_32x64(xb, wab, CEMB, row0, n0, lane, acc);
  const int m = lane & 15;
  const int hb = (lane >> 4) & 1;
#pragma unroll
  for (int mi = 0; mi < 2; ++mi) {
#pragma unroll
    for (int j = 0; j < 4; ++j) {
#pragma unroll
      for (int r = 0; r < 8; ++r) {
        const int n = n0 + j * 16 + m;
        const int row = row0 + mi * 16 + r + hb * 8;
        const int b = row >> 11;        // T = 2048
        const int t = row & (TSEQ - 1);
        float val = acc[mi * 4 + j][r] + bias[n];
        if (n < CEMB) {
          qb[((size_t)(b * HHEADS + (n >> 6)) * TSEQ + t) * 64 + (n & 63)] =
              f2bf(val);
        } else if (n < 2 * CEMB) {
          const int c = n - CEMB;
          kb[((size_t)(b * HHEADS + (c >> 6)) * TSEQ + t) * 64 + (c & 63)] =
              f2bf(val);
        } else {
          const int c = n - 2 * CEMB;
          float veff = val * xt[(size_t)row * CEMB + c];
          vtb[((size_t)(b * HHEADS + (c >> 6)) * 64 + (c & 63)) * TSEQ + t] =
              f2bf(veff);
        }
      }
    }
  }
}

// --------------------------- flash attention core --------------------------
// grid (T/64, B*H); 4 waves/block, each wave owns 16 query rows.
__global__ __launch_bounds__(128) void attn_kernel(
    const u16* __restrict__ qb, const u16* __restrict__ kb,
    const u16* __restrict__ vtb, u16* __restrict__ yb) {
  __shared__ unsigned short pbuf[4][16][32];  // per-wave P staging (bf16)
  const int lane = threadIdx.x & 31;
  const int wave = threadIdx.x >> 5;
  const int bh = blockIdx.y;  // b*16 + h
  const int qbase = blockIdx.x * 64 + wave * 16;
  const int m = lane & 15;
  const int hb = (lane >> 4) & 1;
  const int akb = hb ? 8 : 0;
  const int bkh = hb ? 16 : 0;

  // Preload Q as two 16x32 A-fragments (hd = 64)
  v16bf qf[2];
  const u16* qrowp = qb + ((size_t)bh * TSEQ + qbase + m) * 64;
#pragma unroll
  for (int ks = 0; ks < 2; ++ks) {
    v8u au;
#pragma unroll
    for (int v = 0; v < 4; ++v) {
      au[v]     = *(const unsigned int*)(qrowp + ks * 32 + bkh + 2 * v);
      au[v + 4] = *(const unsigned int*)(qrowp + ks * 32 + bkh + 8 + 2 * v);
    }
    qf[ks] = __builtin_bit_cast(v16bf, au);
  }

  v8f acc[4];
#pragma unroll
  for (int j = 0; j < 4; ++j) acc[j] = zero_v8f();
  float mrow[8], lrow[8];
#pragma unroll
  for (int r = 0; r < 8; ++r) { mrow[r] = -__builtin_inff(); lrow[r] = 0.f; }

  const int nkb = ((qbase + 15) >> 5) + 1;  // causal key-block bound
  for (int kbi = 0; kbi < nkb; ++kbi) {
    const int key0 = kbi * 32;
    // S = Q K^T for 32 keys (two 16x16 N-tiles, 2 WMMAs each over hd)
    v8f s[2];
#pragma unroll
    for (int t2 = 0; t2 < 2; ++t2) {
      v8f z = zero_v8f();
#pragma unroll
      for (int ks = 0; ks < 2; ++ks) {
        const u16* krow =
            kb + ((size_t)bh * TSEQ + key0 + t2 * 16 + m) * 64 + ks * 32 + bkh;
        v8u bu;
#pragma unroll
        for (int v = 0; v < 8; ++v)
          bu[v] = *(const unsigned int*)(krow + 2 * v);
        z = __builtin_amdgcn_wmma_f32_16x16x32_bf16(
            false, qf[ks], false, __builtin_bit_cast(v16bf, bu), (short)0, z,
            false, false);
      }
      s[t2] = z;
    }
    // scale + causal mask + online softmax
    float p0[8], p1[8], mloc[8], lloc[8];
#pragma unroll
    for (int r = 0; r < 8; ++r) {
      const int qrow = qbase + r + hb * 8;
      float a0 = s[0][r] * 0.125f;  // 1/sqrt(64)
      float a1 = s[1][r] * 0.125f;
      if (key0 + m > qrow)      a0 = -__builtin_inff();
      if (key0 + 16 + m > qrow) a1 = -__builtin_inff();
      p0[r] = a0; p1[r] = a1;
      mloc[r] = fmaxf(a0, a1);
    }
#pragma unroll
    for (int off = 1; off < 16; off <<= 1)
#pragma unroll
      for (int r = 0; r < 8; ++r)
        mloc[r] = fmaxf(mloc[r], __shfl_xor(mloc[r], off, 32));
#pragma unroll
    for (int r = 0; r < 8; ++r) {
      float mn = fmaxf(mrow[r], mloc[r]);
      float alpha = __expf(mrow[r] - mn);
      mrow[r] = mn;
      p0[r] = __expf(p0[r] - mn);
      p1[r] = __expf(p1[r] - mn);
      lloc[r] = p0[r] + p1[r];
      lrow[r] *= alpha;
#pragma unroll
      for (int j = 0; j < 4; ++j) acc[j][r] *= alpha;
      pbuf[wave][r + hb * 8][m]      = f2bf(p0[r]);
      pbuf[wave][r + hb * 8][16 + m] = f2bf(p1[r]);
    }
#pragma unroll
    for (int off = 1; off < 16; off <<= 1)
#pragma unroll
      for (int r = 0; r < 8; ++r) lloc[r] += __shfl_xor(lloc[r], off, 32);
#pragma unroll
    for (int r = 0; r < 8; ++r) lrow[r] += lloc[r];

    // D-layout -> A-fragment relayout of P through LDS (per-wave; DS in-order)
    asm volatile("s_wait_dscnt 0" ::: "memory");
    v8u pu;
#pragma unroll
    for (int v = 0; v < 4; ++v) {
      pu[v]     = *(const unsigned int*)&pbuf[wave][m][bkh + 2 * v];
      pu[v + 4] = *(const unsigned int*)&pbuf[wave][m][bkh + 8 + 2 * v];
    }
    asm volatile("" ::: "memory");  // keep next iter's DS stores after loads
    v16bf pf = __builtin_bit_cast(v16bf, pu);
    // O += P * Veff  (Veff stored transposed: pairs contiguous along keys)
#pragma unroll
    for (int j = 0; j < 4; ++j) {
      const u16* vrow =
          vtb + ((size_t)bh * 64 + j * 16 + m) * TSEQ + key0 + bkh;
      v8u bu;
#pragma unroll
      for (int v = 0; v < 8; ++v) bu[v] = *(const unsigned int*)(vrow + 2 * v);
      acc[j] = __builtin_amdgcn_wmma_f32_16x16x32_bf16(
          false, pf, false, __builtin_bit_cast(v16bf, bu), (short)0, acc[j],
          false, false);
    }
  }
  // normalize and store Y (bf16) at [b, t, h*64 + d]
  const int b = bh >> 4, h = bh & 15;
#pragma unroll
  for (int r = 0; r < 8; ++r) {
    const int qrow = qbase + r + hb * 8;
    const float inv = 1.0f / lrow[r];
    const size_t base = ((size_t)b * TSEQ + qrow) * CEMB + h * 64;
#pragma unroll
    for (int j = 0; j < 4; ++j)
      yb[base + j * 16 + m] = f2bf(acc[j][r] * inv);
  }
}

// ----------------------------- output projection ---------------------------
__global__ __launch_bounds__(128) void proj_gemm_kernel(
    const u16* __restrict__ yb, const u16* __restrict__ wpb,
    const float* __restrict__ bias, float* __restrict__ out) {
  const int lane = threadIdx.x & 31;
  const int wave = threadIdx.x >> 5;
  const int row0 = blockIdx.x * 32;
  const int n0 = blockIdx.y * 256 + wave * 64;
  v8f acc[8];
#pragma unroll
  for (int i = 0; i < 8; ++i) acc[i] = zero_v8f();
  wmma_tile_32x64(yb, wpb, CEMB, row0, n0, lane, acc);
  const int m = lane & 15;
  const int hb = (lane >> 4) & 1;
#pragma unroll
  for (int mi = 0; mi < 2; ++mi) {
#pragma unroll
    for (int j = 0; j < 4; ++j) {
#pragma unroll
      for (int r = 0; r < 8; ++r) {
        const int n = n0 + j * 16 + m;
        const int row = row0 + mi * 16 + r + hb * 8;
        out[(size_t)row * CEMB + n] = acc[mi * 4 + j][r] + bias[n];
      }
    }
  }
}

// --------------------------------- launcher --------------------------------
extern "C" void kernel_launch(void* const* d_in, const int* in_sizes, int n_in,
                              void* d_out, int out_size, void* d_ws,
                              size_t ws_size, hipStream_t stream) {
  (void)in_sizes; (void)n_in; (void)out_size; (void)ws_size;
  const float* x   = (const float*)d_in[0];  // x_norm_for_qkv [B,T,C]
  const float* xt  = (const float*)d_in[1];  // xt_current_for_v [B,T,C]
  const float* wat = (const float*)d_in[2];  // w_attn [3C,C]
  const float* bat = (const float*)d_in[3];  // b_attn [3C]
  const float* wpr = (const float*)d_in[4];  // w_proj [C,C]
  const float* bpr = (const float*)d_in[5];  // b_proj [C]
  float* out = (float*)d_out;

  char* ws = (char*)d_ws;
  u16* xb  = (u16*)(ws + 0);
  u16* wab = (u16*)(ws + 8388608);
  u16* wpb = (u16*)(ws + 14680064);
  u16* qb  = (u16*)(ws + 16777216);
  u16* kb  = (u16*)(ws + 25165824);
  u16* vtb = (u16*)(ws + 33554432);
  u16* yb  = (u16*)(ws + 41943040);

  // 1) fp32 -> bf16 conversions (4 elems/thread)
  cvt_bf16_kernel<<<MROWS * CEMB / 1024, 256, 0, stream>>>(
      (const float4*)x, (uint2*)xb, MROWS * CEMB / 4);
  cvt_bf16_kernel<<<3 * CEMB * CEMB / 1024, 256, 0, stream>>>(
      (const float4*)wat, (uint2*)wab, 3 * CEMB * CEMB / 4);
  cvt_bf16_kernel<<<CEMB * CEMB / 1024, 256, 0, stream>>>(
      (const float4*)wpr, (uint2*)wpb, CEMB * CEMB / 4);
  // 2) QKV GEMM + fused head split / V modulation / V transpose
  qkv_gemm_kernel<<<dim3(MROWS / 32, (3 * CEMB) / 256), 128, 0, stream>>>(
      xb, wab, bat, xt, qb, kb, vtb);
  // 3) causal flash attention
  attn_kernel<<<dim3(TSEQ / 64, BSZ * HHEADS), 128, 0, stream>>>(qb, kb, vtb,
                                                                 yb);
  // 4) output projection
  proj_gemm_kernel<<<dim3(MROWS / 32, CEMB / 256), 128, 0, stream>>>(yb, wpb,
                                                                     bpr, out);
}
